// LSTMnet_57380763074600
// MI455X (gfx1250) — matrix operand order repaired and checked by reading, not compiled, not run on hardware
//
#include <hip/hip_runtime.h>

#define B_ 64
#define S_ 1024
#define D_ 256
#define H_ 256
#define O_ 2
#define HP 264   // padded LDS pitch in halves (528B = 132 dwords -> 4-bank skew, 16B aligned)

typedef __attribute__((ext_vector_type(16))) _Float16 v16h;
typedef __attribute__((ext_vector_type(8)))  _Float16 v8h;
typedef __attribute__((ext_vector_type(8)))  float    v8f;

// ---- WMMA helper: D = A(16x32 f16) * B(32x16 f16) + C(16x16 f32) ----
static __device__ inline v8f wmma_f16(v16h a, v16h b, v8f c) {
  return __builtin_amdgcn_wmma_f32_16x16x32_f16(
      /*neg_a=*/false, a, /*neg_b=*/false, b,
      /*c_mod=*/(short)0, c, /*reuse_a=*/false, /*reuse_b=*/false);
}

// A-tile from row-major f32: p0 -> 8 floats (k..k+7), p1 -> 8 floats (k+16..k+23)
static __device__ inline v16h load_a_f32(const float* __restrict__ p0,
                                         const float* __restrict__ p1) {
  v16h a;
#pragma unroll
  for (int i = 0; i < 8; ++i) a[i] = (_Float16)p0[i];
#pragma unroll
  for (int i = 0; i < 8; ++i) a[8 + i] = (_Float16)p1[i];
  return a;
}

// A-tile from f16 (16B-aligned): two b128 loads
static __device__ inline v16h load_a_f16(const _Float16* p0, const _Float16* p1) {
  v8h lo = *(const v8h*)p0;
  v8h hi = *(const v8h*)p1;
  v16h a;
#pragma unroll
  for (int i = 0; i < 8; ++i) { a[i] = lo[i]; a[8 + i] = hi[i]; }
  return a;
}

// B-tile from f16: halves k..k+7 and k+16..k+23 per lane (two b128 loads)
static __device__ inline v16h load_b_f16(const _Float16* p) {
  v8h lo = *(const v8h*)p;
  v8h hi = *(const v8h*)(p + 16);
  v16h b;
#pragma unroll
  for (int i = 0; i < 8; ++i) { b[i] = lo[i]; b[8 + i] = hi[i]; }
  return b;
}

// fast gates: v_rcp_f32 instead of IEEE divide sequences
static __device__ inline float fast_sigmoid(float a) {
  return __builtin_amdgcn_rcpf(1.0f + __expf(-a));
}
static __device__ inline float fast_tanh(float a) {
  const float e2 = __expf(2.0f * a);
  return (e2 - 1.0f) * __builtin_amdgcn_rcpf(e2 + 1.0f);
}

// =====================================================================
// Phase 0: Wx f32 -> f16 (row-major [H][D]) so GEMM B-tiles need no cvt
// =====================================================================
__global__ __launch_bounds__(256) void cvt_wx(const float* __restrict__ Wx,
                                              _Float16* __restrict__ Wx16) {
  const int i = blockIdx.x * 256 + threadIdx.x;
  Wx16[i] = (_Float16)Wx[i];
}

// =====================================================================
// Phase 1: xproj[s*B+b][h] = x[b][s][:] . Wx[h][:] + bx[h]   (f32 out)
// grid.x = (B*S)/16 row tiles, block = 128 threads (4 waves),
// wave w covers n-tiles 4w..4w+3. Per k-step: preload 4 B-tiles in one
// clause, then 4 back-to-back independent WMMAs. Epilogue stores use one
// base pointer + constant immediate offsets (stride v*B*H*4 = v*64KB).
// =====================================================================
__global__ __launch_bounds__(128) void xproj_gemm(const float* __restrict__ x,
                                                  const _Float16* __restrict__ Wx16,
                                                  const float* __restrict__ bx,
                                                  float* __restrict__ xproj) {
  const int lane = threadIdx.x & 31;
  const int wave = threadIdx.x >> 5;          // 0..3
  const int r0 = blockIdx.x * 16;             // row tile within (B*S); same b for all 16 rows
  const int mA  = lane & 15;
  const int kbA = (lane & 16) ? 8 : 0;
  const int nB  = lane & 15;
  const int kbB = (lane & 16) ? 8 : 0;

  v8f acc[4] = {};

  const float* xrow = x + (size_t)(r0 + mA) * D_ + kbA;
  const _Float16* wbase = Wx16 + (size_t)(wave * 64 + nB) * D_ + kbB;
  for (int kk = 0; kk < D_; kk += 32) {
    v16h A = load_a_f32(xrow + kk, xrow + kk + 16);
    v16h Bt[4];
#pragma unroll
    for (int t = 0; t < 4; ++t)
      Bt[t] = load_b_f16(wbase + (size_t)t * 16 * D_ + kk);   // const offsets -> one clause
#pragma unroll
    for (int t = 0; t < 4; ++t)
      acc[t] = wmma_f16(A, Bt[t], acc[t]);
  }

  const int mofs = (lane & 16) ? 8 : 0;
  const int b  = r0 >> 10;                    // batch index (constant per block)
  const int s0 = r0 & (S_ - 1);               // first timestep of this row tile
  float* obase = xproj + ((size_t)(s0 + mofs) * B_ + b) * H_ + wave * 64 + nB;
#pragma unroll
  for (int t = 0; t < 4; ++t) {
    const float bias = bx[wave * 64 + t * 16 + nB];
    float* op = obase + t * 16;
#pragma unroll
    for (int v = 0; v < 8; ++v)               // byte offset v*B*H*4: immediate
      op[(size_t)v * (B_ * H_)] = acc[t][v] + bias;
  }
}

// =====================================================================
// Phase 2+3: sequential LSTM scan on ONE workgroup (32 waves, one WGP).
// Wh (f16) + h (f16) resident in LDS; c in registers (tile ownership is
// static per wave). Per step: 2 tiles/wave x 8 K-steps of WMMA, gates
// with v_rcp_f32, h write, O=2 output projection (vector LDS reads +
// shfl reduce), and a global_prefetch of the next step's xproj slice.
// =====================================================================
__global__ __launch_bounds__(1024) void lstm_scan(const float* __restrict__ Wh,
                                                  const float* __restrict__ bh,
                                                  const float* __restrict__ Wo,
                                                  const float* __restrict__ bo,
                                                  const float* __restrict__ xproj,
                                                  float* __restrict__ out) {
  extern __shared__ __align__(16) char smem[];
  _Float16* whl = (_Float16*)smem;              // [H_][HP]  f16  (135168 B)
  _Float16* hl  = whl + (size_t)H_ * HP;        // [B_][HP]  f16  (33792 B)
  float* bhl    = (float*)(hl + (size_t)B_ * HP);
  float* wol    = bhl + H_;
  float* bol    = wol + O_ * H_;

  const int tid = threadIdx.x;

  // Stage weights / init state in LDS
  for (int i = tid; i < H_ * H_; i += 1024) {
    const int n = i >> 8, k = i & 255;
    whl[n * HP + k] = (_Float16)Wh[i];
  }
  for (int i = tid; i < B_ * HP; i += 1024) hl[i] = (_Float16)0.0f;
  if (tid < H_)      bhl[tid] = bh[tid];
  if (tid < O_ * H_) wol[tid] = Wo[tid];
  if (tid < O_)      bol[tid] = bo[tid];
  __syncthreads();

  const int lane = tid & 31;
  const int wave = tid >> 5;        // 0..31
  const int nt  = wave >> 1;        // 0..15 -> n-tile
  const int m0  = ((wave & 1) * 2) * 16;
  const int m1  = m0 + 16;

  const int mA  = lane & 15;
  const int kbA = (lane & 16) ? 8 : 0;
  const int nB  = lane & 15;
  const int kbB = (lane & 16) ? 8 : 0;
  const int n0  = nt * 16;

  const _Float16* h0row = hl + (m0 + mA) * HP + kbA;
  const _Float16* h1row = hl + (m1 + mA) * HP + kbA;
  const _Float16* wrow  = whl + (n0 + nB) * HP + kbB;

  const int mofs = (lane & 16) ? 8 : 0;
  const int nn   = n0 + (lane & 15);
  const float bhn = bhl[nn];

  v8f c0 = {}, c1 = {};

  // Output-projection assignment: 128 (b,o) pairs x 8 partial lanes
  const int pair = tid >> 3;        // 0..127
  const int pb   = pair >> 1;       // batch
  const int po   = pair & 1;        // output channel
  const int pk0  = (tid & 7) * 32;
  const float bias_o = bol[po];
  const v8h*   hvec = (const v8h*)(hl + pb * HP + pk0);   // 16B aligned
  const float* wvec = wol + po * H_ + pk0;

  // Running xproj base: one 64-bit pointer, all 16 loads use immediate offsets
  const float* xp = xproj + (size_t)(m0 + mofs) * H_ + nn;
  const char*  pfp = (const char*)(xproj + (size_t)B_ * H_) + (size_t)tid * 64;
  float* outp = out + (size_t)pb * S_ * O_ + po;

  for (int s = 0; s < S_; ++s) {
    // Prefetch next step's xproj slice (64KB spread over 1024 threads)
    if (s + 1 < S_) __builtin_prefetch(pfp, 0, 0);

    v8f a0 = {}, a1 = {};
#pragma unroll
    for (int kk = 0; kk < H_; kk += 32) {
      v16h Bt = load_b_f16(wrow + kk);                 // shared across both m-tiles
      v16h A0 = load_a_f16(h0row + kk, h0row + kk + 16);
      v16h A1 = load_a_f16(h1row + kk, h1row + kk + 16);
      a0 = wmma_f16(A0, Bt, a0);
      a1 = wmma_f16(A1, Bt, a1);
    }

    _Float16 hn0[8], hn1[8];
#pragma unroll
    for (int v = 0; v < 8; ++v) {                      // xp[v*H_]: immediate offsets
      const float a  = a0[v] + bhn + xp[v * H_];
      const float sg = fast_sigmoid(a);
      const float c  = sg * (c0[v] + fast_tanh(a));    // sg*c + sg*g
      c0[v] = c;
      hn0[v] = (_Float16)(sg * fast_tanh(c));
    }
#pragma unroll
    for (int v = 0; v < 8; ++v) {                      // second m-tile: +16 rows
      const float a  = a1[v] + bhn + xp[(16 + v) * H_];
      const float sg = fast_sigmoid(a);
      const float c  = sg * (c1[v] + fast_tanh(a));
      c1[v] = c;
      hn1[v] = (_Float16)(sg * fast_tanh(c));
    }
    xp  += B_ * H_;
    pfp += (size_t)B_ * H_ * sizeof(float);

    __syncthreads();   // everyone finished reading h(s-1)
#pragma unroll
    for (int v = 0; v < 8; ++v) {
      hl[(m0 + mofs + v) * HP + nn] = hn0[v];
      hl[(m1 + mofs + v) * HP + nn] = hn1[v];
    }
    __syncthreads();   // h(s) visible to all waves

    // out[b*S+s][o] = h(s)[b][:] . Wo[o][:] + bo[o]  (vector LDS reads)
    float psum = 0.0f;
#pragma unroll
    for (int j = 0; j < 4; ++j) {
      v8h hv = hvec[j];                                // ds_load_b128
#pragma unroll
      for (int i = 0; i < 8; ++i)
        psum += (float)hv[i] * wvec[j * 8 + i];
    }
    psum += __shfl_xor(psum, 1);
    psum += __shfl_xor(psum, 2);
    psum += __shfl_xor(psum, 4);
    if ((tid & 7) == 0) outp[(size_t)s * O_] = psum + bias_o;
  }
}

extern "C" void kernel_launch(void* const* d_in, const int* in_sizes, int n_in,
                              void* d_out, int out_size, void* d_ws, size_t ws_size,
                              hipStream_t stream) {
  const float* x  = (const float*)d_in[0];
  const float* Wh = (const float*)d_in[1];
  const float* bh = (const float*)d_in[2];
  const float* Wx = (const float*)d_in[3];
  const float* bx = (const float*)d_in[4];
  const float* Wo = (const float*)d_in[5];
  const float* bo = (const float*)d_in[6];
  float* out   = (float*)d_out;

  float* xproj      = (float*)d_ws;                               // S*B*H f32 = 64 MiB
  _Float16* wx16    = (_Float16*)(xproj + (size_t)S_ * B_ * H_);  // 128 KiB

  // Phase 0: one-shot f16 conversion of Wx (removes cvt from GEMM hot loop)
  cvt_wx<<<(H_ * D_) / 256, 256, 0, stream>>>(Wx, wx16);

  // Phase 1: big parallel WMMA GEMM over all timesteps
  xproj_gemm<<<(B_ * S_) / 16, 128, 0, stream>>>(x, wx16, bx, xproj);

  // Phase 2+3: single-WGP persistent recurrence with LDS-resident weights
  const size_t smem = (size_t)(H_ * HP + B_ * HP) * sizeof(_Float16) +
                      (size_t)(H_ + O_ * H_ + O_) * sizeof(float);
  lstm_scan<<<1, 1024, smem, stream>>>(Wh, bh, Wo, bo, xproj, out);
}
